// MoE_21406117003466
// MI455X (gfx1250) — matrix operand order repaired and checked by reading, not compiled, not run on hardware
//
#include <hip/hip_runtime.h>
#include <hip/hip_bf16.h>

#define E_    8
#define D_    1024
#define H_    4096
#define NTOK  8192                       // B*S
#define MAX_SLOTS (NTOK * 2 + E_ * 64)   // 16896 (top-2 routing + per-expert pad to 64)
#define MAX_TILES (MAX_SLOTS / 64)       // 264

typedef __attribute__((ext_vector_type(16))) __bf16       v16bf;
typedef __attribute__((ext_vector_type(8)))  float        v8f;
typedef __attribute__((ext_vector_type(8)))  unsigned int v8u;
typedef __attribute__((ext_vector_type(4)))  unsigned int u32x4;
typedef __attribute__((ext_vector_type(8)))  int          i32x8;
typedef __attribute__((ext_vector_type(4)))  int          i32x4;

static __device__ __forceinline__ unsigned short f2bfbits(float f) {
    union { float f; unsigned u; } v; v.f = f;
    unsigned r = v.u + 0x7FFFu + ((v.u >> 16) & 1u);   // round-to-nearest-even
    return (unsigned short)(r >> 16);
}

// 16 contiguous bf16, 32B-aligned -> clean ds_load_b128 pair
static __device__ __forceinline__ v16bf loadfragB(const unsigned short* p) {
    return __builtin_bit_cast(v16bf, *(const v8u*)p);
}
// A fragment: shorts [0..7] and [16..23] from a 16B-aligned base
static __device__ __forceinline__ v16bf loadfragA(const unsigned short* p) {
    uint4 lo = *(const uint4*)p;
    uint4 hi = *(const uint4*)(p + 16);
    v8u v; v[0] = lo.x; v[1] = lo.y; v[2] = lo.z; v[3] = lo.w;
           v[4] = hi.x; v[5] = hi.y; v[6] = hi.z; v[7] = hi.w;
    return __builtin_bit_cast(v16bf, v);
}

// ---------------------------------------------------------------- zero
__global__ void zero_kernel(float* out, int n4, int* cnt) {
    int i = blockIdx.x * blockDim.x + threadIdx.x;
    if (i < n4) ((float4*)out)[i] = make_float4(0.f, 0.f, 0.f, 0.f);
    if (i < E_) cnt[i] = 0;
}

// ---------------------------------------------------------------- router
__global__ void router_kernel(const float* __restrict__ x, const float* __restrict__ Wr,
                              const float* __restrict__ br,
                              int* cnt, int* idx, float* wgt) {
    int tok = blockIdx.x * blockDim.x + threadIdx.x;
    if (tok >= NTOK) return;
    const float* xr = x + (size_t)tok * D_;
    float acc[E_];
    #pragma unroll
    for (int e = 0; e < E_; ++e) acc[e] = br[e];
    for (int d = 0; d < D_; d += 4) {
        float4 xv = *(const float4*)(xr + d);
        #pragma unroll
        for (int q = 0; q < 4; ++q) {
            const float* wr = Wr + (size_t)(d + q) * E_;
            float xs = (&xv.x)[q];
            #pragma unroll
            for (int e = 0; e < E_; ++e) acc[e] += xs * wr[e];
        }
    }
    int i0 = 0; float l0 = acc[0];
    #pragma unroll
    for (int e = 1; e < E_; ++e) if (acc[e] > l0) { l0 = acc[e]; i0 = e; }
    int i1 = -1; float l1 = -3.0e38f;
    #pragma unroll
    for (int e = 0; e < E_; ++e) if (e != i0 && acc[e] > l1) { l1 = acc[e]; i1 = e; }
    float t  = __expf(l1 - l0);
    float w0 = 1.f / (1.f + t);
    float w1 = t / (1.f + t);
    int s0 = atomicAdd(&cnt[i0], 1);
    idx[i0 * NTOK + s0] = tok; wgt[i0 * NTOK + s0] = w0;
    int s1 = atomicAdd(&cnt[i1], 1);
    idx[i1 * NTOK + s1] = tok; wgt[i1 * NTOK + s1] = w1;
}

// ---------------------------------------------------------------- scan (pad to 64, prefix sum)
__global__ void scan_kernel(const int* cnt, int* off, int* idx, float* wgt) {
    if (blockIdx.x == 0 && threadIdx.x == 0) {
        int acc = 0;
        for (int e = 0; e < E_; ++e) {
            off[e] = acc;
            int c  = cnt[e];
            int cp = (c + 63) & ~63;
            for (int s = c; s < cp; ++s) { idx[e * NTOK + s] = -1; wgt[e * NTOK + s] = 0.f; }
            acc += cp;
        }
        off[E_] = acc;
    }
}

// expert lookup (M-tile = 64 -> shift 6)
static __device__ __forceinline__ void tile_lookup(const int* off, int t, int* sMeta) {
    int tot = off[E_] >> 6;
    int e = 0, lt = 0, oe = 0;
    if (t < tot) {
        for (int k = E_ - 1; k >= 0; --k) if (t >= (off[k] >> 6)) { e = k; break; }
        oe = off[e];
        lt = t - (oe >> 6);
    }
    sMeta[0] = e; sMeta[1] = lt; sMeta[2] = oe; sMeta[3] = tot;
}

// ---------------------------------------------------------------- h = silu(x Wg) * (x Wu)
// Block: 128 thr (4 waves). Tile: 64 rows x 64 cols. Wave w -> cols w*16..+15, 4 M-subtiles.
__global__ void __launch_bounds__(128)
gemm_h_kernel(const float* __restrict__ x, const float* __restrict__ Wg,
              const float* __restrict__ Wu,
              const int* __restrict__ off, const int* __restrict__ idx,
              unsigned short* __restrict__ hbuf) {
    __shared__ __align__(32) unsigned short aT[64 * 32];     // [M][K]
    __shared__ __align__(32) unsigned short bgT[64 * 32];    // [N][K] transposed
    __shared__ __align__(32) unsigned short buT[64 * 32];
    __shared__ int sid[64];
    __shared__ int sMeta[4];

    int t = blockIdx.x;
    if (threadIdx.x == 0) tile_lookup(off, t, sMeta);
    __syncthreads();
    if (t >= sMeta[3]) return;                 // uniform exit
    int e = sMeta[0], lt = sMeta[1], offE = sMeta[2];
    if (threadIdx.x < 64) sid[threadIdx.x] = idx[e * NTOK + lt * 64 + threadIdx.x];
    __syncthreads();

    const float* Wge = Wg + (size_t)e * D_ * H_;
    const float* Wue = Wu + (size_t)e * D_ * H_;
    int colbase = blockIdx.y * 64;             // within H

    int tid   = threadIdx.x;
    int lane  = tid & 31;
    int wave  = tid >> 5;
    int laneM = lane & 15;
    int akb   = (lane < 16) ? 0 : 8;           // A: K base per lane half
    int bkb   = (lane < 16) ? 0 : 8;           // B: K base per lane half

    v8f accg[4] = {{}, {}, {}, {}};
    v8f accu[4] = {{}, {}, {}, {}};

    for (int kb = 0; kb < D_; kb += 32) {
        // A tile 64x32: fp32 x -> bf16 LDS (padded rows -> 0). 512 float4 / 128 thr.
        #pragma unroll
        for (int it = 0; it < 4; ++it) {
            int q = it * 128 + tid;
            int r = q >> 3, c = (q & 7) * 4;
            int tok = sid[r];
            float4 xv = make_float4(0.f, 0.f, 0.f, 0.f);
            if (tok >= 0) xv = *(const float4*)(x + (size_t)tok * D_ + kb + c);
            unsigned p0 = (unsigned)f2bfbits(xv.x) | ((unsigned)f2bfbits(xv.y) << 16);
            unsigned p1 = (unsigned)f2bfbits(xv.z) | ((unsigned)f2bfbits(xv.w) << 16);
            *(uint2*)&aT[r * 32 + c] = make_uint2(p0, p1);
        }
        // B tiles 32x64 (gate & up), stored [N][K]. 512 float4 each / 128 thr.
        #pragma unroll
        for (int it = 0; it < 4; ++it) {
            int q = it * 128 + tid;
            int r = q >> 4, c = (q & 15) * 4;  // r = K row, c = N col
            float4 g = *(const float4*)(Wge + (size_t)(kb + r) * H_ + colbase + c);
            float4 u = *(const float4*)(Wue + (size_t)(kb + r) * H_ + colbase + c);
            #pragma unroll
            for (int j = 0; j < 4; ++j) {
                bgT[(c + j) * 32 + r] = f2bfbits((&g.x)[j]);
                buT[(c + j) * 32 + r] = f2bfbits((&u.x)[j]);
            }
        }
        __syncthreads();

        v16bf a0 = loadfragA(&aT[(0  + laneM) * 32 + akb]);
        v16bf a1 = loadfragA(&aT[(16 + laneM) * 32 + akb]);
        v16bf a2 = loadfragA(&aT[(32 + laneM) * 32 + akb]);
        v16bf a3 = loadfragA(&aT[(48 + laneM) * 32 + akb]);
        v16bf bg = loadfragA(&bgT[(wave * 16 + laneM) * 32 + bkb]);
        v16bf bu = loadfragA(&buT[(wave * 16 + laneM) * 32 + bkb]);

        accg[0] = __builtin_amdgcn_wmma_f32_16x16x32_bf16(false, a0, false, bg, (short)0, accg[0], false, false);
        accg[1] = __builtin_amdgcn_wmma_f32_16x16x32_bf16(false, a1, false, bg, (short)0, accg[1], false, false);
        accg[2] = __builtin_amdgcn_wmma_f32_16x16x32_bf16(false, a2, false, bg, (short)0, accg[2], false, false);
        accg[3] = __builtin_amdgcn_wmma_f32_16x16x32_bf16(false, a3, false, bg, (short)0, accg[3], false, false);
        accu[0] = __builtin_amdgcn_wmma_f32_16x16x32_bf16(false, a0, false, bu, (short)0, accu[0], false, false);
        accu[1] = __builtin_amdgcn_wmma_f32_16x16x32_bf16(false, a1, false, bu, (short)0, accu[1], false, false);
        accu[2] = __builtin_amdgcn_wmma_f32_16x16x32_bf16(false, a2, false, bu, (short)0, accu[2], false, false);
        accu[3] = __builtin_amdgcn_wmma_f32_16x16x32_bf16(false, a3, false, bu, (short)0, accu[3], false, false);
        __syncthreads();
    }

    int rowbase = offE + lt * 64;
    int mofs = (lane < 16) ? 0 : 8;
    int col  = colbase + wave * 16 + laneM;
    #pragma unroll
    for (int ms = 0; ms < 4; ++ms) {
        #pragma unroll
        for (int v = 0; v < 8; ++v) {
            int M = ms * 16 + v + mofs;
            float g = accg[ms][v], u = accu[ms][v];
            float h = g * (1.f / (1.f + __expf(-g))) * u;
            hbuf[(size_t)(rowbase + M) * H_ + col] = f2bfbits(h);
        }
    }
}

// ---------------------------------------------------------------- out += w * (h Wd)
// Block: 128 thr (4 waves). Tile: 64x64 of out. A tile streamed with the Tensor Data Mover.
__global__ void __launch_bounds__(128)
gemm_out_kernel(const unsigned short* __restrict__ hbuf, const float* __restrict__ Wd,
                const int* __restrict__ off, const int* __restrict__ idx,
                const float* __restrict__ wgt, float* __restrict__ out) {
    __shared__ __align__(32) unsigned short aT[64 * 32];     // [M][K] (bf16, filled by TDM)
    __shared__ __align__(32) unsigned short bT[64 * 32];     // [N][K]
    __shared__ int   sid[64];
    __shared__ float sw[64];
    __shared__ int sMeta[4];

    int t = blockIdx.x;
    if (threadIdx.x == 0) tile_lookup(off, t, sMeta);
    __syncthreads();
    if (t >= sMeta[3]) return;
    int e = sMeta[0], lt = sMeta[1], offE = sMeta[2];
    if (threadIdx.x < 64) {
        sid[threadIdx.x] = idx[e * NTOK + lt * 64 + threadIdx.x];
        sw[threadIdx.x]  = wgt[e * NTOK + lt * 64 + threadIdx.x];
    }
    __syncthreads();

    const float* Wde = Wd + (size_t)e * H_ * D_;
    int colbase = blockIdx.y * 64;             // within D

    int tid   = threadIdx.x;
    int lane  = tid & 31;
    int wave  = tid >> 5;
    int laneM = lane & 15;
    int akb   = (lane < 16) ? 0 : 8;
    int bkb   = (lane < 16) ? 0 : 8;
    int rowbase = offE + lt * 64;
    unsigned ldsA = (unsigned)(uintptr_t)aT;   // low 32b of flat LDS addr = wave-relative offset

    v8f acc[4] = {{}, {}, {}, {}};

    for (int kb = 0; kb < H_; kb += 32) {
        // A tile 64x32 bf16 from hbuf via TDM: 2-D tile, 32 elems/row (2B), row stride H.
        if (wave == 0) {                       // uniform wave-level branch; TDM ignores EXEC
            unsigned long long ga =
                (unsigned long long)(uintptr_t)(hbuf + (size_t)rowbase * H_ + kb);
            u32x4 g0;
            g0[0] = 1u;                                    // count=1, user D#
            g0[1] = ldsA;                                  // lds_addr (bytes)
            g0[2] = (unsigned)ga;                          // global_addr[31:0]
            g0[3] = (unsigned)(ga >> 32) | 0x80000000u;    // global_addr[56:32] | type=2
            i32x8 g1;
            g1[0] = 0x00010000;                            // data_size=2B; no multicast/pad
            g1[1] = (int)(32u << 16);                      // tensor_dim0 = 32
            g1[2] = (int)(64u << 16);                      // tensor_dim1 = 64
            g1[3] = (int)(32u << 16);                      // tile_dim0 = 32
            g1[4] = 64;                                    // tile_dim1 = 64 (tile_dim2 = 0)
            g1[5] = 4096;                                  // tensor_dim0_stride = H
            g1[6] = 0; g1[7] = 0;
            i32x4 z4 = {0, 0, 0, 0};                       // groups 2/3: dims unused
            i32x8 z8 = {0, 0, 0, 0, 0, 0, 0, 0};
            __builtin_amdgcn_tensor_load_to_lds(g0, g1, z4, z4, z8, 0);
            __builtin_amdgcn_s_wait_tensorcnt((short)0);
        }
        // B tile 32x64 fp32 Wd -> bf16 [N][K]
        #pragma unroll
        for (int it = 0; it < 4; ++it) {
            int q = it * 128 + tid;
            int r = q >> 4, c = (q & 15) * 4;
            float4 w = *(const float4*)(Wde + (size_t)(kb + r) * D_ + colbase + c);
            #pragma unroll
            for (int j = 0; j < 4; ++j) bT[(c + j) * 32 + r] = f2bfbits((&w.x)[j]);
        }
        __syncthreads();

        v16bf a0 = loadfragA(&aT[(0  + laneM) * 32 + akb]);
        v16bf a1 = loadfragA(&aT[(16 + laneM) * 32 + akb]);
        v16bf a2 = loadfragA(&aT[(32 + laneM) * 32 + akb]);
        v16bf a3 = loadfragA(&aT[(48 + laneM) * 32 + akb]);
        v16bf b  = loadfragA(&bT[(wave * 16 + laneM) * 32 + bkb]);

        acc[0] = __builtin_amdgcn_wmma_f32_16x16x32_bf16(false, a0, false, b, (short)0, acc[0], false, false);
        acc[1] = __builtin_amdgcn_wmma_f32_16x16x32_bf16(false, a1, false, b, (short)0, acc[1], false, false);
        acc[2] = __builtin_amdgcn_wmma_f32_16x16x32_bf16(false, a2, false, b, (short)0, acc[2], false, false);
        acc[3] = __builtin_amdgcn_wmma_f32_16x16x32_bf16(false, a3, false, b, (short)0, acc[3], false, false);
        __syncthreads();
    }

    int mofs = (lane < 16) ? 0 : 8;
    int col  = colbase + wave * 16 + laneM;
    #pragma unroll
    for (int ms = 0; ms < 4; ++ms) {
        #pragma unroll
        for (int v = 0; v < 8; ++v) {
            int Ml = ms * 16 + v + mofs;
            int tok = sid[Ml];
            if (tok >= 0) {
                // exactly 2 commutative adds per element onto zero base -> deterministic
                atomicAdd(out + (size_t)tok * D_ + col, acc[ms][v] * sw[Ml]);
            }
        }
    }
}

// ---------------------------------------------------------------- launch
extern "C" void kernel_launch(void* const* d_in, const int* in_sizes, int n_in,
                              void* d_out, int out_size, void* d_ws, size_t ws_size,
                              hipStream_t stream) {
    const float* x  = (const float*)d_in[0];
    const float* Wr = (const float*)d_in[1];
    const float* br = (const float*)d_in[2];
    const float* Wg = (const float*)d_in[3];
    const float* Wu = (const float*)d_in[4];
    const float* Wd = (const float*)d_in[5];
    float* out = (float*)d_out;

    char* ws = (char*)d_ws;
    int*   cnt = (int*)(ws);                      // 8 ints
    int*   off = (int*)(ws + 64);                 // 9 ints
    int*   idx = (int*)(ws + 128);                // 8*8192 ints  (256 KB)
    float* wgt = (float*)(ws + 128 + 262144);     // 8*8192 floats (256 KB)
    unsigned short* hbuf = (unsigned short*)(ws + 524544);  // MAX_SLOTS*H bf16 (~138 MB)

    zero_kernel<<<(NTOK * D_ / 4 + 255) / 256, 256, 0, stream>>>(out, NTOK * D_ / 4, cnt);
    router_kernel<<<NTOK / 256, 256, 0, stream>>>(x, Wr, br, cnt, idx, wgt);
    scan_kernel<<<1, 32, 0, stream>>>(cnt, off, idx, wgt);
    dim3 gh(MAX_TILES, H_ / 64);
    gemm_h_kernel<<<gh, 128, 0, stream>>>(x, Wg, Wu, off, idx, hbuf);
    dim3 go(MAX_TILES, D_ / 64);
    gemm_out_kernel<<<go, 128, 0, stream>>>(hbuf, Wd, off, idx, wgt, out);
}
